// FSEL_11433202942085
// MI455X (gfx1250) — matrix operand order repaired and checked by reading, not compiled, not run on hardware
//
#include <hip/hip_runtime.h>
#include <hip/hip_bf16.h>
#include <math.h>

// ---------------------------------------------------------------------------
// CDNA5 (gfx1250) implementation of the "former" block.
// All dense math (1x1 convs, channel-attention Gram/apply, and every FFT --
// recast as DFT-matrix GEMMs since 96-pt DFT and 9216=96*96 four-step map
// perfectly onto 16x16x32 bf16 WMMA) runs on the matrix pipes.
// LDS tiles are stored in WMMA fragment order so B-fragment loads are
// 2x ds_load_b128; A fragments come as 4x global_load_b128 (float4).
// ---------------------------------------------------------------------------

typedef __attribute__((ext_vector_type(16))) __bf16 v16bf;
typedef __attribute__((ext_vector_type(4)))  __bf16 v4bf;
typedef __attribute__((ext_vector_type(8)))  float  v8f;

#define PI2F 6.28318530717958647692f

// ---------------------------------------------------------------------------
// LayerNorm over channel dim of NCHW. One wave32 per pixel.
// ---------------------------------------------------------------------------
__global__ void k_ln_ch(const float* __restrict__ x, const float* __restrict__ g,
                        const float* __restrict__ b, float* __restrict__ y,
                        int C, long HW, long CHW) {
  int wave = threadIdx.x >> 5, lane = threadIdx.x & 31;
  long pix = (long)blockIdx.x * (blockDim.x >> 5) + wave;
  long base = (pix / HW) * CHW + (pix % HW);
  float s = 0.f, ss = 0.f;
  for (int c = lane; c < C; c += 32) {
    float v = x[base + (long)c * HW];
    s += v; ss += v * v;
  }
  for (int m = 16; m; m >>= 1) { s += __shfl_xor(s, m, 32); ss += __shfl_xor(ss, m, 32); }
  float mean = s / (float)C;
  float var  = ss / (float)C - mean * mean;
  float inv  = rsqrtf(var + 1e-5f);
  for (int c = lane; c < C; c += 32) {
    float v = x[base + (long)c * HW];
    y[base + (long)c * HW] = (v - mean) * inv * g[c] + b[c];
  }
}

// Build one lane's A fragment (16x32 bf16, ISA layout) from row-major W rows.
// Requires k0+32 <= K and 16B-aligned rows (all our K are multiples of 4).
__device__ inline v16bf load_a_frag_fast(const float* __restrict__ row, int k0, int kb) {
  const float* pa = row + k0 + kb;
  float4 w0 = *(const float4*)(pa);
  float4 w1 = *(const float4*)(pa + 4);
  float4 w2 = *(const float4*)(pa + 16);
  float4 w3 = *(const float4*)(pa + 20);
  v16bf a;
  a[0]  = (__bf16)w0.x; a[1]  = (__bf16)w0.y; a[2]  = (__bf16)w0.z; a[3]  = (__bf16)w0.w;
  a[4]  = (__bf16)w1.x; a[5]  = (__bf16)w1.y; a[6]  = (__bf16)w1.z; a[7]  = (__bf16)w1.w;
  a[8]  = (__bf16)w2.x; a[9]  = (__bf16)w2.y; a[10] = (__bf16)w2.z; a[11] = (__bf16)w2.w;
  a[12] = (__bf16)w3.x; a[13] = (__bf16)w3.y; a[14] = (__bf16)w3.z; a[15] = (__bf16)w3.w;
  return a;
}

// ---------------------------------------------------------------------------
// Generic batched GEMM, D = alpha * W(MxK) * X(KxN) [+ bias] [+ res], via
// V_WMMA_F32_16X16X32_BF16.  Block = 128 threads = 4 waves, each wave owns a
// 16x16 D tile (64 rows x 16 cols per block).  X tile staged f32->bf16 in LDS
// in fragment order: sBf[lane][e] with lane=((k&8)<<1)+n, e=(k&7)|((k&16)>>1).
// XMODE 0: X[k,n]=Xz[k*ldx+n]   (row-major K)
// XMODE 1: X[k,n]=Xz[n*ldx+k]   (col-major K, used for Q*K^T)
// Per-z base offsets: off = (z/zInner)*Out + (z%zInner)*In  for W, X, D.
// K loop is split: full 32-wide tiles (branch-free) + one guarded tail.
// ---------------------------------------------------------------------------
template <int XMODE>
__global__ void k_gemm_t(const float* __restrict__ Wt, const float* __restrict__ X,
                         const float* __restrict__ bias, const float* res,
                         float* D, float alpha,
                         int M, int K, int N, int ldx, int zInner,
                         long wOut, long wIn, long xOut, long xIn, long dOut, long dIn) {
  const int wave = threadIdx.x >> 5;
  const int lane = threadIdx.x & 31;
  const int n0 = blockIdx.x * 16;
  const int m0 = blockIdx.y * 64 + wave * 16;
  const int z  = blockIdx.z;
  const float* Wz = Wt + (long)(z / zInner) * wOut + (long)(z % zInner) * wIn;
  const float* Xz = X  + (long)(z / zInner) * xOut + (long)(z % zInner) * xIn;
  float*       Dz = D  + (long)(z / zInner) * dOut + (long)(z % zInner) * dIn;
  const float* Rz = res ? res + (long)(z / zInner) * dOut + (long)(z % zInner) * dIn : nullptr;

  __shared__ alignas(32) __bf16 sBf[32][16];
  v8f acc = {};
  int mA = m0 + (lane & 15); if (mA >= M) mA = M - 1;   // clamp; discarded at store
  const int kb = (lane >> 4) * 8;

  // Staging coordinates for this thread (blockDim.x == 128 always).
  const int snn = threadIdx.x & 15;
  const int skk = (threadIdx.x >> 4) * 4;
  __bf16* sdst = &sBf[((skk & 8) << 1) + snn][(skk & 7) | ((skk & 16) >> 1)];

  const int kFull = K & ~31;
  for (int k0 = 0; k0 < kFull; k0 += 32) {
    {
      float4 v;
      if (XMODE == 0) {
        const float* p = Xz + (long)(k0 + skk) * ldx + (n0 + snn);
        v.x = p[0]; v.y = p[ldx]; v.z = p[2 * (long)ldx]; v.w = p[3 * (long)ldx];
      } else {
        v = *(const float4*)(Xz + (long)(n0 + snn) * ldx + (k0 + skk));
      }
      v4bf pk;
      pk[0] = (__bf16)v.x; pk[1] = (__bf16)v.y; pk[2] = (__bf16)v.z; pk[3] = (__bf16)v.w;
      *(v4bf*)sdst = pk;
    }
    __syncthreads();
    if (k0 + 32 < K) {  // gfx1250 global_prefetch_b8 for next tile
      const float* pf = (XMODE == 0) ? &Xz[(long)(k0 + 32) * ldx + n0]
                                     : &Xz[(long)n0 * ldx + (k0 + 32)];
      __builtin_prefetch(pf, 0, 0);
    }
    v16bf a = load_a_frag_fast(Wz + (long)mA * K, k0, kb);
    v16bf bfr = *(const v16bf*)&sBf[lane][0];           // 2x ds_load_b128
    acc = __builtin_amdgcn_wmma_f32_16x16x32_bf16(false, a, false, bfr,
                                                  (short)0, acc, false, false);
    __syncthreads();
  }
  if (kFull < K) {  // guarded tail (K not a multiple of 32, e.g. K=16)
    const int k0 = kFull;
    for (int e = threadIdx.x; e < 512; e += 128) {
      int kk = e >> 4, nn = e & 15;
      int k = k0 + kk;
      float v = 0.f;
      if (k < K) v = (XMODE == 0) ? Xz[(long)k * ldx + (n0 + nn)]
                                  : Xz[(long)(n0 + nn) * ldx + k];
      sBf[((kk & 8) << 1) + nn][(kk & 7) | ((kk & 16) >> 1)] = (__bf16)v;
    }
    __syncthreads();
    v16bf a;
    #pragma unroll
    for (int e = 0; e < 16; ++e) {
      int koff = (e < 8) ? (kb + e) : (16 + kb + (e - 8));
      int k = k0 + koff;
      float wv = (k < K) ? Wz[(long)mA * K + k] : 0.f;
      a[e] = (__bf16)wv;
    }
    v16bf bfr = *(const v16bf*)&sBf[lane][0];
    acc = __builtin_amdgcn_wmma_f32_16x16x32_bf16(false, a, false, bfr,
                                                  (short)0, acc, false, false);
    __syncthreads();
  }
  // Epilogue: C/D layout -> row m0 + i + 8*(lane/16), col n0 + lane%16.
  {
    const int nn = lane & 15;
    const int mr = (lane >> 4) * 8;
    #pragma unroll
    for (int i = 0; i < 8; ++i) {
      int m = m0 + mr + i;
      if (m < M) {
        long off = (long)m * N + (n0 + nn);
        float v = alpha * acc[i];
        if (bias) v += bias[m];
        if (Rz)   v += Rz[off];
        Dz[off] = v;
      }
    }
  }
}

// ---------------------------------------------------------------------------
// Complex DFT of length L (<=96, multiple of 32 here) along an arbitrary
// strided axis, as GEMM: Y = F * X, 4 bf16 WMMAs per tile.  bf16 WMMA has no
// A/B negate modifier (NEG={CNeg,0,0} per ISA), so the -Fi*Xi term uses a
// VALU-negated copy of the Fi fragment.  One block owns ceil(L/16) row-tiles
// (all of F's rows) for one 16-column slab; in-place safe.
// Element addressing: src[(n/nInner)*outS + (n%nInner)*inS + k*kS]
// Xi may be null (real input).  nTotal must be a multiple of 16.
// ---------------------------------------------------------------------------
__global__ void k_cdft(const float* __restrict__ Fr, const float* __restrict__ Fi, int L,
                       const float* __restrict__ Xr, const float* __restrict__ Xi,
                       float* Yr, float* Yi,
                       long nTotal, long nInner, long inS, long outS, long kS,
                       float scale) {
  const int wave = threadIdx.x >> 5;
  const int lane = threadIdx.x & 31;
  const int m0 = wave * 16;
  const long n0 = (long)blockIdx.x * 16;

  __shared__ alignas(32) __bf16 sRf[32][16];
  __shared__ alignas(32) __bf16 sIf[32][16];
  v8f accR = {}, accI = {};
  int mA = m0 + (lane & 15); if (mA >= L) mA = L - 1;
  const int kb = (lane >> 4) * 8;

  for (int k0 = 0; k0 < L; k0 += 32) {   // L % 32 == 0 for L in {32, 96}
    for (int t = threadIdx.x; t < 128; t += blockDim.x) {
      int nn = t & 15, kq = t >> 4, kk = kq * 4;
      long n = n0 + nn;
      long base = (n / nInner) * outS + (n % nInner) * inS + (long)(k0 + kk) * kS;
      float4 vr, vi;
      if (kS == 1) {
        vr = *(const float4*)(Xr + base);
        if (Xi) vi = *(const float4*)(Xi + base);
        else    vi.x = vi.y = vi.z = vi.w = 0.f;
      } else {
        vr.x = Xr[base]; vr.y = Xr[base + kS]; vr.z = Xr[base + 2 * kS]; vr.w = Xr[base + 3 * kS];
        if (Xi) { vi.x = Xi[base]; vi.y = Xi[base + kS]; vi.z = Xi[base + 2 * kS]; vi.w = Xi[base + 3 * kS]; }
        else    vi.x = vi.y = vi.z = vi.w = 0.f;
      }
      v4bf pr, pi;
      pr[0] = (__bf16)vr.x; pr[1] = (__bf16)vr.y; pr[2] = (__bf16)vr.z; pr[3] = (__bf16)vr.w;
      pi[0] = (__bf16)vi.x; pi[1] = (__bf16)vi.y; pi[2] = (__bf16)vi.z; pi[3] = (__bf16)vi.w;
      *(v4bf*)&sRf[((kk & 8) << 1) + nn][(kk & 7) | ((kk & 16) >> 1)] = pr;
      *(v4bf*)&sIf[((kk & 8) << 1) + nn][(kk & 7) | ((kk & 16) >> 1)] = pi;
    }
    __syncthreads();
    v16bf aR = load_a_frag_fast(Fr + (long)mA * L, k0, kb);
    v16bf aI = load_a_frag_fast(Fi + (long)mA * L, k0, kb);
    v16bf aIn;
    #pragma unroll
    for (int e = 0; e < 16; ++e) aIn[e] = (__bf16)(-(float)aI[e]);
    v16bf bR = *(const v16bf*)&sRf[lane][0];
    v16bf bI = *(const v16bf*)&sIf[lane][0];
    accR = __builtin_amdgcn_wmma_f32_16x16x32_bf16(false, aR,  false, bR, (short)0, accR, false, false);
    accR = __builtin_amdgcn_wmma_f32_16x16x32_bf16(false, aIn, false, bI, (short)0, accR, false, false);
    accI = __builtin_amdgcn_wmma_f32_16x16x32_bf16(false, aR,  false, bI, (short)0, accI, false, false);
    accI = __builtin_amdgcn_wmma_f32_16x16x32_bf16(false, aI,  false, bR, (short)0, accI, false, false);
    __syncthreads();
  }
  {
    const int nn = lane & 15;
    const int mr = (lane >> 4) * 8;
    #pragma unroll
    for (int i = 0; i < 8; ++i) {
      int u = m0 + mr + i;
      long n = n0 + nn;
      if (u < L) {
        long off = (n / nInner) * outS + (n % nInner) * inS + (long)u * kS;
        Yr[off] = accR[i] * scale;
        Yi[off] = accI[i] * scale;
      }
    }
  }
}

// ---------------------------------------------------------------------------
// DFT table init: F[u,k] = exp(sign*2*pi*i*u*k/L).
// ---------------------------------------------------------------------------
__global__ void k_init_dft(float* Fr, float* Fi, int L, float sign) {
  int i = blockIdx.x * blockDim.x + threadIdx.x;
  if (i < L * L) {
    int u = i / L, k = i % L;
    float ang = sign * PI2F * (float)((u * k) % L) / (float)L;
    Fr[i] = cosf(ang);
    Fi[i] = sinf(ang);
  }
}

// Four-step FFT twiddle: element [row][a][b] *= exp(sign*2*pi*i*a*b/(Na*Nb))
__global__ void k_ctwiddle(float* r, float* im, long rows, int Na, int Nb, float sign) {
  long idx = (long)blockIdx.x * blockDim.x + threadIdx.x;
  long total = rows * (long)Na * Nb;
  if (idx >= total) return;
  int b = (int)(idx % Nb);
  int a = (int)((idx / Nb) % Na);
  long N = (long)Na * Nb;
  float ang = sign * PI2F * (float)(((long)a * b) % N) / (float)N;
  float cr = cosf(ang), ci = sinf(ang);
  float x = r[idx], y = im[idx];
  r[idx]  = x * cr - y * ci;
  im[idx] = x * ci + y * cr;
}

// y = act(x*s + t), per-channel s,t (nullable). act: 0 none, 1 relu, 2 sigmoid
__global__ void k_chan_act(const float* __restrict__ x, const float* __restrict__ s,
                           const float* __restrict__ t, float* __restrict__ y,
                           int C, long HW, long total, int act) {
  long idx = (long)blockIdx.x * blockDim.x + threadIdx.x;
  if (idx >= total) return;
  int c = (int)((idx / HW) % C);
  float v = x[idx];
  if (s) v = v * s[c] + t[c];
  if (act == 1) v = fmaxf(v, 0.f);
  else if (act == 2) v = 1.f / (1.f + __expf(-v));
  y[idx] = v;
}

// complex *= real gate (in place)
__global__ void k_cgate(float* r, float* im, const float* __restrict__ g, long total) {
  long idx = (long)blockIdx.x * blockDim.x + threadIdx.x;
  if (idx >= total) return;
  float w = g[idx];
  r[idx] *= w; im[idx] *= w;
}

// out[b, ocOff+c, p] = |r + i*im|   (embeds into a concat buffer)
__global__ void k_cabs(const float* __restrict__ r, const float* __restrict__ im,
                       float* __restrict__ out, int C, int CoutTot, int ocOff,
                       long HW, long total) {
  long idx = (long)blockIdx.x * blockDim.x + threadIdx.x;
  if (idx >= total) return;
  long p = idx % HW;
  int  c = (int)((idx / HW) % C);
  long b = idx / (HW * C);
  float x = r[idx], y = im[idx];
  out[(b * CoutTot + ocOff + c) * HW + p] = sqrtf(x * x + y * y);
}

// out[b, ocOff+c, p] = gelu(v)*v  (exact erf gelu; embeds into concat buffer)
__global__ void k_gelugate(const float* __restrict__ in, float* __restrict__ out,
                           int C, int CoutTot, int ocOff, long HW, long total) {
  long idx = (long)blockIdx.x * blockDim.x + threadIdx.x;
  if (idx >= total) return;
  long p = idx % HW;
  int  c = (int)((idx / HW) % C);
  long b = idx / (HW * C);
  float v = in[idx];
  float gel = 0.5f * v * (1.f + erff(v * 0.70710678118f));
  out[(b * CoutTot + ocOff + c) * HW + p] = gel * v;
}

// out = a + b (+ c)
__global__ void k_add3(const float* __restrict__ a, const float* __restrict__ b,
                       const float* __restrict__ c, float* __restrict__ out, long total) {
  long idx = (long)blockIdx.x * blockDim.x + threadIdx.x;
  if (idx >= total) return;
  float v = a[idx] + b[idx];
  if (c) v += c[idx];
  out[idx] = v;
}

// In-place L2 normalize rows of length L (complex if im != null).
// base = (row/rowsPerB)*batchStride + (row%rowsPerB)*L
__global__ void k_l2norm(float* r, float* im, long L, int rowsPerB, long batchStride) {
  long row = blockIdx.x;
  long base = (row / rowsPerB) * batchStride + (row % rowsPerB) * L;
  float* pr = r + base;
  float* pi = im ? im + base : nullptr;
  float s = 0.f;
  for (long t = threadIdx.x; t < L; t += blockDim.x) {
    float a = pr[t]; s += a * a;
    if (pi) { float c = pi[t]; s += c * c; }
  }
  for (int m = 16; m; m >>= 1) s += __shfl_xor(s, m, 32);
  __shared__ float red[8];
  if ((threadIdx.x & 31) == 0) red[threadIdx.x >> 5] = s;
  __syncthreads();
  if (threadIdx.x == 0) {
    float tot = 0.f;
    for (int w = 0; w < (int)(blockDim.x >> 5); ++w) tot += red[w];
    red[0] = tot;
  }
  __syncthreads();
  float inv = 1.f / fmaxf(sqrtf(red[0]), 1e-12f);
  for (long t = threadIdx.x; t < L; t += blockDim.x) {
    pr[t] *= inv;
    if (pi) pi[t] *= inv;
  }
}

// Softmax over last dim (=32) of [rows,32]; wave per row; scale by temp[head].
__global__ void k_softmax32(const float* __restrict__ in, float* __restrict__ out,
                            const float* __restrict__ temp, long rows, int NH) {
  int wave = threadIdx.x >> 5, lane = threadIdx.x & 31;
  long row = (long)blockIdx.x * (blockDim.x >> 5) + wave;
  if (row >= rows) return;
  float ts = temp[(row / 32) % NH];
  float v = in[row * 32 + lane] * ts;
  float mx = v;
  for (int m = 16; m; m >>= 1) mx = fmaxf(mx, __shfl_xor(mx, m, 32));
  float e = __expf(v - mx);
  float s = e;
  for (int m = 16; m; m >>= 1) s += __shfl_xor(s, m, 32);
  out[row * 32 + lane] = e / s;
}

// Generic grouped conv (3x3 / 5x5 depthwise-ish). Embeds into concat buffer.
__global__ void k_dwconv(const float* __restrict__ x, const float* __restrict__ w,
                         const float* __restrict__ bias, float* __restrict__ y,
                         int B, int Cin, int Cout, int H, int Wd, int ks, int pad,
                         int groups, int CoutTot, int ocOff) {
  long idx = (long)blockIdx.x * blockDim.x + threadIdx.x;
  long total = (long)B * Cout * H * Wd;
  if (idx >= total) return;
  int xw = (int)(idx % Wd);
  long t = idx / Wd;
  int yh = (int)(t % H); t /= H;
  int oc = (int)(t % Cout);
  int bb = (int)(t / Cout);
  int icPer = Cin / groups, ocPer = Cout / groups;
  int grp = oc / ocPer, icb = grp * icPer;
  float acc = bias ? bias[oc] : 0.f;
  for (int ic = 0; ic < icPer; ++ic)
    for (int ky = 0; ky < ks; ++ky) {
      int iy = yh + ky - pad;
      if (iy < 0 || iy >= H) continue;
      for (int kx = 0; kx < ks; ++kx) {
        int ix = xw + kx - pad;
        if (ix < 0 || ix >= Wd) continue;
        acc += w[(((long)oc * icPer + ic) * ks + ky) * ks + kx] *
               x[(((long)bb * Cin + icb + ic) * H + iy) * Wd + ix];
      }
    }
  y[(((long)bb * CoutTot + ocOff + oc) * H + yh) * Wd + xw] = acc;
}

// ---------------------------------------------------------------------------
// Host orchestration
// ---------------------------------------------------------------------------
extern "C" void kernel_launch(void* const* d_in, const int* in_sizes, int n_in,
                              void* d_out, int out_size, void* d_ws, size_t ws_size,
                              hipStream_t stream) {
  (void)in_sizes; (void)n_in; (void)out_size; (void)ws_size;
  const int B = 8, C = 256, H = 96, Wd = 96, NH = 8;
  const long HW = (long)H * Wd;           // 9216
  const long U = (long)B * C * HW;        // one [B,C,H,W] f32 tensor

  int i = 0;
  const float* x        = (const float*)d_in[i++];
  const float* ln0_g    = (const float*)d_in[i++];
  const float* ln0_b    = (const float*)d_in[i++];
  const float* ln1_g    = (const float*)d_in[i++];
  const float* ln1_b    = (const float*)d_in[i++];
  const float* pos_w    = (const float*)d_in[i++];
  const float* pos_b    = (const float*)d_in[i++];
  const float* qd3_w    = (const float*)d_in[i++];
  const float* qd3_b    = (const float*)d_in[i++];
  const float* qd5_w    = (const float*)d_in[i++];
  const float* qd5_b    = (const float*)d_in[i++];
  const float* ssa_temp = (const float*)d_in[i++];
  const float* d3_w     = (const float*)d_in[i++];
  const float* d3_b     = (const float*)d_in[i++];
  const float* d5_w     = (const float*)d_in[i++];
  const float* d5_b     = (const float*)d_in[i++];
  const float* ssa_pw   = (const float*)d_in[i++];
  const float* fsa_temp = (const float*)d_in[i++];
  const float* fsa_w1   = (const float*)d_in[i++];
  const float* fsa_s1   = (const float*)d_in[i++];
  const float* fsa_t1   = (const float*)d_in[i++];
  const float* fsa_w2   = (const float*)d_in[i++];
  const float* fsa_b2   = (const float*)d_in[i++];
  const float* fsa_pw   = (const float*)d_in[i++];
  const float* f0_w1    = (const float*)d_in[i++];
  const float* f0_s1    = (const float*)d_in[i++];
  const float* f0_t1    = (const float*)d_in[i++];
  const float* f0_w2    = (const float*)d_in[i++];
  const float* f0_b2    = (const float*)d_in[i++];
  const float* d0_w     = (const float*)d_in[i++];
  const float* d0_b     = (const float*)d_in[i++];
  const float* f1_w1    = (const float*)d_in[i++];
  const float* f1_s1    = (const float*)d_in[i++];
  const float* f1_t1    = (const float*)d_in[i++];
  const float* f1_w2    = (const float*)d_in[i++];
  const float* f1_b2    = (const float*)d_in[i++];
  const float* d1_w     = (const float*)d_in[i++];
  const float* d1_b     = (const float*)d_in[i++];
  const float* proj_w   = (const float*)d_in[i++];
  float* out = (float*)d_out;

  char* wsb = (char*)d_ws;
  size_t off = 0;
  auto alloc = [&](long nf) -> float* {
    float* p = (float*)(wsb + off);
    off += (size_t)nf * sizeof(float);
    off = (off + 255) & ~(size_t)255;
    return p;
  };

  auto gemm = [&](const float* Wt, const float* X, const float* bias, const float* res,
                  float* D, float alpha, int M, int K, int N, int ldx, int xmode,
                  int nz, int zInner, long wOut, long wIn, long xOut, long xIn,
                  long dOut, long dIn) {
    dim3 g(N / 16, (M + 63) / 64, nz);
    if (xmode == 0)
      k_gemm_t<0><<<g, 128, 0, stream>>>(Wt, X, bias, res, D, alpha, M, K, N, ldx,
                                         zInner, wOut, wIn, xOut, xIn, dOut, dIn);
    else
      k_gemm_t<1><<<g, 128, 0, stream>>>(Wt, X, bias, res, D, alpha, M, K, N, ldx,
                                         zInner, wOut, wIn, xOut, xIn, dOut, dIn);
  };
  auto cdft = [&](const float* Fr, const float* Fi, int L, const float* Xr,
                  const float* Xi, float* Yr, float* Yi, long nTot, long nInner,
                  long inS, long outS, long kS, float scale) {
    int waves = (L + 15) / 16;
    dim3 g((unsigned)((nTot + 15) / 16));
    k_cdft<<<g, 32 * waves, 0, stream>>>(Fr, Fi, L, Xr, Xi, Yr, Yi, nTot, nInner,
                                         inS, outS, kS, scale);
  };
  auto ew = [&](long total) { return dim3((unsigned)((total + 255) / 256)); };

  // -------- DFT tables (L2 resident) --------
  float* F96fr = alloc(9216); float* F96fi = alloc(9216);
  float* F96ir = alloc(9216); float* F96ii = alloc(9216);
  float* F32ir = alloc(1024); float* F32ii = alloc(1024);
  k_init_dft<<<ew(9216), 256, 0, stream>>>(F96fr, F96fi, 96, -1.f);
  k_init_dft<<<ew(9216), 256, 0, stream>>>(F96ir, F96ii, 96, 1.f);
  k_init_dft<<<ew(1024), 256, 0, stream>>>(F32ir, F32ii, 32, 1.f);

  // -------- x0 = LN0(x) --------
  float* x0 = alloc(U);
  k_ln_ch<<<B * HW / 8, 256, 0, stream>>>(x, ln0_g, ln0_b, x0, C, HW, (long)C * HW);

  // ======================= SSA =======================
  float* h3 = alloc(3 * U);   // 1x1 C->3C
  gemm(pos_w, x0, pos_b, nullptr, h3, 1.f, 3 * C, C, (int)HW, (int)HW, 0,
       B, B, 0, 0, 0, (long)C * HW, 0, 3L * C * HW);
  float* qkv = alloc(3 * U);  // [qd3 | qd5] grouped convs
  {
    long t = (long)B * 384 * HW;
    k_dwconv<<<ew(t), 256, 0, stream>>>(h3, qd3_w, qd3_b, qkv, B, 768, 384, H, Wd, 3, 1, 384, 768, 0);
    k_dwconv<<<ew(t), 256, 0, stream>>>(h3, qd5_w, qd5_b, qkv, B, 768, 384, H, Wd, 5, 2, 384, 768, 384);
  }
  // l2-normalize q (ch 0..255) and k (ch 256..511) rows over HW
  k_l2norm<<<B * 256, 256, 0, stream>>>(qkv, nullptr, HW, 256, 768L * HW);
  k_l2norm<<<B * 256, 256, 0, stream>>>((float*)(qkv + 256 * HW), nullptr, HW, 256, 768L * HW);
  // S = q k^T  (per (b,h): 32x32, K=9216)
  float* Ssa = alloc((long)B * NH * 1024);
  float* Asa = alloc((long)B * NH * 1024);
  gemm(qkv, qkv + 256 * HW, nullptr, nullptr, Ssa, 1.f, 32, (int)HW, 32, (int)HW, 1,
       B * NH, NH, 768L * HW, 32L * HW, 768L * HW, 32L * HW, (long)NH * 1024, 1024);
  k_softmax32<<<(B * NH * 32) / 8, 256, 0, stream>>>(Ssa, Asa, ssa_temp, (long)B * NH * 32, NH);
  // attn = A @ v  -> concat[ch 0..255]; d3/d5(x0) -> ch 256..511
  float* cat2a = alloc(2 * U);
  gemm(Asa, qkv + 512 * HW, nullptr, nullptr, cat2a, 1.f, 32, 32, (int)HW, (int)HW, 0,
       B * NH, NH, (long)NH * 1024, 1024, 768L * HW, 32L * HW, 512L * HW, 32L * HW);
  {
    long t = (long)B * 128 * HW;
    k_dwconv<<<ew(t), 256, 0, stream>>>(x0, d3_w, d3_b, cat2a, B, 256, 128, H, Wd, 3, 1, 128, 512, 256);
    k_dwconv<<<ew(t), 256, 0, stream>>>(x0, d5_w, d5_b, cat2a, B, 256, 128, H, Wd, 5, 2, 128, 512, 384);
  }
  float* ssaOut = alloc(U);
  gemm(ssa_pw, cat2a, nullptr, nullptr, ssaOut, 1.f, C, 2 * C, (int)HW, (int)HW, 0,
       B, B, 0, 0, 0, 512L * HW, 0, (long)C * HW);

  // ======================= FSA =======================
  float* Xr = alloc(U); float* Xi = alloc(U);
  // fft2(x0): DFT96 along W (contig), then along H (stride 96); in place 2nd pass
  cdft(F96fr, F96fi, 96, x0, nullptr, Xr, Xi, (long)B * C * H, (long)B * C * H, 96, 0, 1, 1.f);
  cdft(F96fr, F96fi, 96, Xr, Xi, Xr, Xi, (long)B * C * 96, 96, 1, 9216, 96, 1.f);
  // qk = l2norm(v) (complex rows over HW) on a copy
  float* Qr = alloc(U); float* Qi = alloc(U);
  (void)hipMemcpyAsync(Qr, Xr, (size_t)U * 4, hipMemcpyDeviceToDevice, stream);
  (void)hipMemcpyAsync(Qi, Xi, (size_t)U * 4, hipMemcpyDeviceToDevice, stream);
  k_l2norm<<<B * C, 256, 0, stream>>>(Qr, Qi, HW, B * C, 0);
  // S = qk qk^T (complex, no conj): Sr = QrQr^T - QiQi^T ; Si = QrQi^T + QiQr^T
  float* Sr = alloc((long)B * NH * 1024); float* Si = alloc((long)B * NH * 1024);
  float* Ar = alloc((long)B * NH * 1024); float* Ai = alloc((long)B * NH * 1024);
  gemm(Qr, Qr, nullptr, nullptr, Sr,  1.f, 32, (int)HW, 32, (int)HW, 1,
       B * NH, NH, (long)C * HW, 32L * HW, (long)C * HW, 32L * HW, (long)NH * 1024, 1024);
  gemm(Qi, Qi, nullptr, Sr,      Sr, -1.f, 32, (int)HW, 32, (int)HW, 1,
       B * NH, NH, (long)C * HW, 32L * HW, (long)C * HW, 32L * HW, (long)NH * 1024, 1024);
  gemm(Qr, Qi, nullptr, nullptr, Si,  1.f, 32, (int)HW, 32, (int)HW, 1,
       B * NH, NH, (long)C * HW, 32L * HW, (long)C * HW, 32L * HW, (long)NH * 1024, 1024);
  gemm(Qi, Qr, nullptr, Si,      Si,  1.f, 32, (int)HW, 32, (int)HW, 1,
       B * NH, NH, (long)C * HW, 32L * HW, (long)C * HW, 32L * HW, (long)NH * 1024, 1024);
  k_softmax32<<<(B * NH * 32) / 8, 256, 0, stream>>>(Sr, Ar, fsa_temp, (long)B * NH * 32, NH);
  k_softmax32<<<(B * NH * 32) / 8, 256, 0, stream>>>(Si, Ai, fsa_temp, (long)B * NH * 32, NH);
  // O = A @ v (complex), v = X heads
  float* Or = alloc(U); float* Oi = alloc(U);
  gemm(Ar, Xr, nullptr, nullptr, Or,  1.f, 32, 32, (int)HW, (int)HW, 0,
       B * NH, NH, (long)NH * 1024, 1024, (long)C * HW, 32L * HW, (long)C * HW, 32L * HW);
  gemm(Ai, Xi, nullptr, Or,      Or, -1.f, 32, 32, (int)HW, (int)HW, 0,
       B * NH, NH, (long)NH * 1024, 1024, (long)C * HW, 32L * HW, (long)C * HW, 32L * HW);
  gemm(Ar, Xi, nullptr, nullptr, Oi,  1.f, 32, 32, (int)HW, (int)HW, 0,
       B * NH, NH, (long)NH * 1024, 1024, (long)C * HW, 32L * HW, (long)C * HW, 32L * HW);
  gemm(Ai, Xr, nullptr, Oi,      Oi,  1.f, 32, 32, (int)HW, (int)HW, 0,
       B * NH, NH, (long)NH * 1024, 1024, (long)C * HW, 32L * HW, (long)C * HW, 32L * HW);
  // ifft2 over head dims [c=32, s=9216]: inv DFT-32 along c, then inv FFT-9216
  // along s via four-step (DFT96 stride-96, twiddle, DFT96 stride-1).
  cdft(F32ir, F32ii, 32, Or, Oi, Or, Oi, (long)B * NH * HW, HW, 1, 32L * HW, HW, 1.f / 32.f);
  cdft(F96ir, F96ii, 96, Or, Oi, Or, Oi, (long)B * C * 96, 96, 1, 9216, 96, 1.f / 96.f);
  k_ctwiddle<<<ew(U), 256, 0, stream>>>(Or, Oi, (long)B * C, 96, 96, 1.f);
  cdft(F96ir, F96ii, 96, Or, Oi, Or, Oi, (long)B * C * 96, (long)B * C * 96, 96, 0, 1, 1.f / 96.f);
  float* cat2f = alloc(2 * U);
  k_cabs<<<ew(U), 256, 0, stream>>>(Or, Oi, cat2f, C, 512, 0, HW, U);
  // frequency gate: sigmoid(conv(relu(bn(conv(X.real))))) applied to X, ifft2, abs
  float* g16 = alloc((long)B * 16 * HW);
  gemm(fsa_w1, Xr, nullptr, nullptr, g16, 1.f, 16, C, (int)HW, (int)HW, 0,
       B, B, 0, 0, 0, (long)C * HW, 0, 16L * HW);
  k_chan_act<<<ew((long)B * 16 * HW), 256, 0, stream>>>(g16, fsa_s1, fsa_t1, g16, 16, HW, (long)B * 16 * HW, 1);
  float* gate = alloc(U);
  gemm(fsa_w2, g16, fsa_b2, nullptr, gate, 1.f, C, 16, (int)HW, (int)HW, 0,
       B, B, 0, 0, 0, 16L * HW, 0, (long)C * HW);
  k_chan_act<<<ew(U), 256, 0, stream>>>(gate, nullptr, nullptr, gate, C, HW, U, 2);
  k_cgate<<<ew(U), 256, 0, stream>>>(Xr, Xi, gate, U);
  cdft(F96ir, F96ii, 96, Xr, Xi, Xr, Xi, (long)B * C * 96, 96, 1, 9216, 96, 1.f / 96.f);
  cdft(F96ir, F96ii, 96, Xr, Xi, Xr, Xi, (long)B * C * 96, (long)B * C * 96, 96, 0, 1, 1.f / 96.f);
  k_cabs<<<ew(U), 256, 0, stream>>>(Xr, Xi, cat2f, C, 512, 256, HW, U);
  float* fsaOut = alloc(U);
  gemm(fsa_pw, cat2f, nullptr, nullptr, fsaOut, 1.f, C, 2 * C, (int)HW, (int)HW, 0,
       B, B, 0, 0, 0, 512L * HW, 0, (long)C * HW);

  // -------- x1 = x + ssa + fsa ; xn = LN1(x1) --------
  float* x1 = alloc(U);
  k_add3<<<ew(U), 256, 0, stream>>>(x, ssaOut, fsaOut, x1, U);
  float* xn = alloc(U);
  k_ln_ch<<<B * HW / 8, 256, 0, stream>>>(x1, ln1_g, ln1_b, xn, C, HW, (long)C * HW);

  // ======================= EFFN =======================
  // f0 = FreqPerception(xn)
  cdft(F96fr, F96fi, 96, xn, nullptr, Xr, Xi, (long)B * C * H, (long)B * C * H, 96, 0, 1, 1.f);
  cdft(F96fr, F96fi, 96, Xr, Xi, Xr, Xi, (long)B * C * 96, 96, 1, 9216, 96, 1.f);
  gemm(f0_w1, Xr, nullptr, nullptr, g16, 1.f, 16, C, (int)HW, (int)HW, 0,
       B, B, 0, 0, 0, (long)C * HW, 0, 16L * HW);
  k_chan_act<<<ew((long)B * 16 * HW), 256, 0, stream>>>(g16, f0_s1, f0_t1, g16, 16, HW, (long)B * 16 * HW, 1);
  gemm(f0_w2, g16, f0_b2, nullptr, gate, 1.f, C, 16, (int)HW, (int)HW, 0,
       B, B, 0, 0, 0, 16L * HW, 0, (long)C * HW);
  k_chan_act<<<ew(U), 256, 0, stream>>>(gate, nullptr, nullptr, gate, C, HW, U, 2);
  k_cgate<<<ew(U), 256, 0, stream>>>(Xr, Xi, gate, U);
  cdft(F96ir, F96ii, 96, Xr, Xi, Xr, Xi, (long)B * C * 96, 96, 1, 9216, 96, 1.f / 96.f);
  cdft(F96ir, F96ii, 96, Xr, Xi, Xr, Xi, (long)B * C * 96, (long)B * C * 96, 96, 0, 1, 1.f / 96.f);
  float* fbuf = alloc(U);
  k_cabs<<<ew(U), 256, 0, stream>>>(Xr, Xi, fbuf, C, C, 0, HW, U);
  // s = depthwise3x3(xn)
  float* sbuf = alloc(U);
  k_dwconv<<<ew(U), 256, 0, stream>>>(xn, d0_w, d0_b, sbuf, B, C, C, H, Wd, 3, 1, C, C, 0);
  // y = [gelu(f)*f, gelu(s)*s]  (2C channels)
  float* y2 = alloc(2 * U);
  k_gelugate<<<ew(U), 256, 0, stream>>>(fbuf, y2, C, 512, 0, HW, U);
  k_gelugate<<<ew(U), 256, 0, stream>>>(sbuf, y2, C, 512, 256, HW, U);
  // f1 = FreqPerception(y) on 2C channels
  float* E2r = alloc(2 * U); float* E2i = alloc(2 * U);
  cdft(F96fr, F96fi, 96, y2, nullptr, E2r, E2i, (long)B * 512 * H, (long)B * 512 * H, 96, 0, 1, 1.f);
  cdft(F96fr, F96fi, 96, E2r, E2i, E2r, E2i, (long)B * 512 * 96, 96, 1, 9216, 96, 1.f);
  float* g16b = alloc((long)B * 16 * HW);
  gemm(f1_w1, E2r, nullptr, nullptr, g16b, 1.f, 16, 512, (int)HW, (int)HW, 0,
       B, B, 0, 0, 0, 512L * HW, 0, 16L * HW);
  k_chan_act<<<ew((long)B * 16 * HW), 256, 0, stream>>>(g16b, f1_s1, f1_t1, g16b, 16, HW, (long)B * 16 * HW, 1);
  float* gate2 = alloc(2 * U);
  gemm(f1_w2, g16b, f1_b2, nullptr, gate2, 1.f, 512, 16, (int)HW, (int)HW, 0,
       B, B, 0, 0, 0, 16L * HW, 0, 512L * HW);
  k_chan_act<<<ew(2 * U), 256, 0, stream>>>(gate2, nullptr, nullptr, gate2, 512, HW, 2 * U, 2);
  k_cgate<<<ew(2 * U), 256, 0, stream>>>(E2r, E2i, gate2, 2 * U);
  cdft(F96ir, F96ii, 96, E2r, E2i, E2r, E2i, (long)B * 512 * 96, 96, 1, 9216, 96, 1.f / 96.f);
  cdft(F96ir, F96ii, 96, E2r, E2i, E2r, E2i, (long)B * 512 * 96, (long)B * 512 * 96, 96, 0, 1, 1.f / 96.f);
  float* cat4 = alloc(4 * U);
  k_cabs<<<ew(2 * U), 256, 0, stream>>>(E2r, E2i, cat4, 512, 1024, 0, HW, 2 * U);
  // s1 = grouped conv3x3 (groups=C on 2C) -> cat4 ch 512..1023
  k_dwconv<<<ew(2 * U), 256, 0, stream>>>(y2, d1_w, d1_b, cat4, B, 512, 512, H, Wd, 3, 1, 256, 1024, 512);
  // proj 4C -> C
  float* effn = alloc(U);
  gemm(proj_w, cat4, nullptr, nullptr, effn, 1.f, C, 4 * C, (int)HW, (int)HW, 0,
       B, B, 0, 0, 0, 1024L * HW, 0, (long)C * HW);

  // -------- out = x1 + effn --------
  k_add3<<<ew(U), 256, 0, stream>>>(x1, effn, nullptr, out, U);
}
